// AdvancedGATModel_90065464197217
// MI455X (gfx1250) — matrix operand order repaired and checked by reading, not compiled, not run on hardware
//
#include <hip/hip_runtime.h>
#include <hip/hip_bf16.h>

// ---------------------------------------------------------------------------
// GATv2 (2 layers, H=4 then H=1) + skips + MLP head for MI455X (gfx1250).
// Dense node transforms use v_wmma_f32_16x16x32_f16 (wave32 WMMA, f32 acc).
// Edge stage uses CSR (built on-device) so softmax + aggregation need no
// float atomics; all per-node feature rows move as coalesced float4 (512B).
// ---------------------------------------------------------------------------

#define NODES   100000
#define INF_    128
#define HIDF    128
#define HEADS   4
#define OUTF    64
#define NEG_SLOPE 0.2f

typedef _Float16 v16h __attribute__((ext_vector_type(16)));
typedef _Float16 v8h  __attribute__((ext_vector_type(8)));
typedef _Float16 v4h  __attribute__((ext_vector_type(4)));
typedef float    v8f  __attribute__((ext_vector_type(8)));

// ---------------------------- small utility kernels ------------------------

__global__ void k_zero_i32(int* p, int n) {
    int i = blockIdx.x * blockDim.x + threadIdx.x;
    if (i < n) p[i] = 0;
}

// float32 -> float16 (4 elems / thread)
__global__ void k_f2h(const float* __restrict__ in, _Float16* __restrict__ out, int n4) {
    int i = blockIdx.x * blockDim.x + threadIdx.x;
    if (i >= n4) return;
    float4 v = ((const float4*)in)[i];
    v4h h = { (_Float16)v.x, (_Float16)v.y, (_Float16)v.z, (_Float16)v.w };
    ((v4h*)out)[i] = h;
}

// W[k][p] (row-major [128 x P], f32) -> Wt[p][k] ([P x 128], f16)
__global__ void k_packw(const float* __restrict__ W, _Float16* __restrict__ Wt, int P) {
    int i = blockIdx.x * blockDim.x + threadIdx.x;
    if (i >= P * 128) return;
    int p = i >> 7, k = i & 127;
    Wt[p * 128 + k] = (_Float16)W[k * P + p];
}

// ------------------------------- CSR build ---------------------------------

__global__ void k_count_deg(const int* __restrict__ dstp, int* __restrict__ deg,
                            int E, int N) {
    int e = blockIdx.x * blockDim.x + threadIdx.x;
    if (e >= E + N) return;
    int d = (e < E) ? dstp[e] : (e - E);   // self loops appended
    atomicAdd(&deg[d], 1);
}

#define SCAN_B 256
__global__ void k_scan_block(const int* __restrict__ deg, int* __restrict__ rowptr,
                             int* __restrict__ partials, int n) {
    __shared__ int sh[SCAN_B];
    int b = blockIdx.x, t = threadIdx.x, i = b * SCAN_B + t;
    int v = (i < n) ? deg[i] : 0;
    sh[t] = v; __syncthreads();
    for (int off = 1; off < SCAN_B; off <<= 1) {
        int x = (t >= off) ? sh[t - off] : 0;
        __syncthreads();
        sh[t] += x; __syncthreads();
    }
    if (i < n) rowptr[i + 1] = sh[t];            // block-local inclusive scan
    if (t == SCAN_B - 1) partials[b] = sh[t];
}

__global__ void k_scan_partials(int* partials, int nb) {     // single block
    __shared__ int sh[1024];
    int t = threadIdx.x;
    sh[t] = (t < nb) ? partials[t] : 0; __syncthreads();
    for (int off = 1; off < 1024; off <<= 1) {
        int x = (t >= off) ? sh[t - off] : 0;
        __syncthreads();
        sh[t] += x; __syncthreads();
    }
    if (t < nb) partials[t] = sh[t];             // inclusive
}

__global__ void k_scan_add(int* rowptr, const int* __restrict__ partials, int n) {
    int i = blockIdx.x * blockDim.x + threadIdx.x;
    if (i == 0) rowptr[0] = 0;
    if (i < n) {
        int b = i / SCAN_B;
        if (b > 0) rowptr[i + 1] += partials[b - 1];
    }
}

__global__ void k_build_csr(const int* __restrict__ srcp, const int* __restrict__ dstp,
                            const int* __restrict__ rowptr, int* __restrict__ fill,
                            int* __restrict__ esrc, int* __restrict__ eidx,
                            int E, int N) {
    int e = blockIdx.x * blockDim.x + threadIdx.x;
    if (e >= E + N) return;
    int s, d;
    if (e < E) { s = srcp[e]; d = dstp[e]; } else { s = e - E; d = s; }
    int pos = rowptr[d] + atomicAdd(&fill[d], 1);
    esrc[pos] = s;
    eidx[pos] = e;
}

// ------------------------------- WMMA GEMM ---------------------------------
// C[n x P] = A_f16[n x 128] @ Bt_f16[P x 128]^T + bias, optional relu,
// optional f16 output. One wave per 16x64 C tile; 16 v_wmma per wave.
__global__ void k_gemm_wmma(const _Float16* __restrict__ A,
                            const _Float16* __restrict__ Bt,
                            const float* __restrict__ bias,
                            float* __restrict__ Cf, _Float16* __restrict__ Ch,
                            int n, int P, int relu) {
    int lane = threadIdx.x & 31;
    int wv   = threadIdx.x >> 5;
    int tile = blockIdx.x * (blockDim.x >> 5) + wv;
    int m0   = tile * 16;
    if (m0 >= n) return;
    int n0 = blockIdx.y * 64;
    int hi = lane >> 4, lo = lane & 15;

    const _Float16* arow = A + (size_t)(m0 + lo) * 128;   // A: M = lane&15
    v8f acc[4] = {};

    #pragma unroll
    for (int kk = 0; kk < 128; kk += 32) {
        union { v16h v; v8h h[2]; } a;
        int k0 = kk + 16 * hi;                   // A frag: K = e + 16*hi
        a.h[0] = *(const v8h*)(arow + k0);
        a.h[1] = *(const v8h*)(arow + k0 + 8);
        #pragma unroll
        for (int t = 0; t < 4; t++) {
            int col = n0 + t * 16 + lo;          // B frag: N = lane&15
            v16h b = *(const v16h*)(Bt + (size_t)col * 128 + kk + 16 * hi);
            acc[t] = __builtin_amdgcn_wmma_f32_16x16x32_f16(
                false, a.v, false, b, (short)0, acc[t], false, false);
        }
    }

    #pragma unroll
    for (int t = 0; t < 4; t++) {
        int col = n0 + t * 16 + lo;
        float bv = bias ? bias[col] : 0.0f;
        #pragma unroll
        for (int r = 0; r < 8; r++) {
            int m = m0 + r + 8 * hi;             // C: M = r + 8*(lane>>4)
            float v = acc[t][r] + bv;
            if (relu) v = fmaxf(v, 0.0f);
            if (Ch) Ch[(size_t)m * P + col] = (_Float16)v;
            else    Cf[(size_t)m * P + col] = v;
        }
    }
}

// ----------------------------- edge logits ---------------------------------
// One wave / edge, each lane owns 4 features (float4). Layer1: 4 heads of 32.
__global__ void k_elog_h4(const float* __restrict__ xl, const float* __restrict__ xr,
                          const int* __restrict__ srcp, const int* __restrict__ dstp,
                          const float* __restrict__ att, float* __restrict__ elog,
                          int E, int N) {
    int e = blockIdx.x * (blockDim.x >> 5) + (threadIdx.x >> 5);
    int lane = threadIdx.x & 31;
    if (e >= E + N) return;
    int s, d;
    if (e < E) { s = srcp[e]; d = dstp[e]; } else { s = e - E; d = s; }
    float4 a = ((const float4*)(xl + (size_t)s * 128))[lane];
    float4 b = ((const float4*)(xr + (size_t)d * 128))[lane];
    float4 t;
    t.x = a.x + b.x; t.y = a.y + b.y; t.z = a.z + b.z; t.w = a.w + b.w;
    t.x = t.x > 0.f ? t.x : NEG_SLOPE * t.x;
    t.y = t.y > 0.f ? t.y : NEG_SLOPE * t.y;
    t.z = t.z > 0.f ? t.z : NEG_SLOPE * t.z;
    t.w = t.w > 0.f ? t.w : NEG_SLOPE * t.w;
    float4 w = ((const float4*)att)[lane];
    float p = t.x * w.x + t.y * w.y + t.z * w.z + t.w * w.w;
    p += __shfl_xor(p, 1); p += __shfl_xor(p, 2); p += __shfl_xor(p, 4);
    if ((lane & 7) == 0) elog[(size_t)e * 4 + (lane >> 3)] = p;
}

// Layer2: single head over all 128 features.
__global__ void k_elog_h1(const float* __restrict__ xl, const float* __restrict__ xr,
                          const int* __restrict__ srcp, const int* __restrict__ dstp,
                          const float* __restrict__ att, float* __restrict__ elog,
                          int E, int N) {
    int e = blockIdx.x * (blockDim.x >> 5) + (threadIdx.x >> 5);
    int lane = threadIdx.x & 31;
    if (e >= E + N) return;
    int s, d;
    if (e < E) { s = srcp[e]; d = dstp[e]; } else { s = e - E; d = s; }
    float4 a = ((const float4*)(xl + (size_t)s * 128))[lane];
    float4 b = ((const float4*)(xr + (size_t)d * 128))[lane];
    float4 t;
    t.x = a.x + b.x; t.y = a.y + b.y; t.z = a.z + b.z; t.w = a.w + b.w;
    t.x = t.x > 0.f ? t.x : NEG_SLOPE * t.x;
    t.y = t.y > 0.f ? t.y : NEG_SLOPE * t.y;
    t.z = t.z > 0.f ? t.z : NEG_SLOPE * t.z;
    t.w = t.w > 0.f ? t.w : NEG_SLOPE * t.w;
    float4 w = ((const float4*)att)[lane];
    float p = t.x * w.x + t.y * w.y + t.z * w.z + t.w * w.w;
    p += __shfl_xor(p, 1); p += __shfl_xor(p, 2); p += __shfl_xor(p, 4);
    p += __shfl_xor(p, 8); p += __shfl_xor(p, 16);
    if (lane == 0) elog[e] = p;
}

// --------------------- per-node softmax + aggregation ----------------------
// One wave / destination node. 4-head variant: lane owns 4 features,
// feature-head = lane>>3; stats computed per head across lanes (lane&3 = head).
__global__ void k_gather_h4(const float* __restrict__ xl, const float* __restrict__ elog,
                            const int* __restrict__ rowptr, const int* __restrict__ esrc,
                            const int* __restrict__ eidx, const float* __restrict__ bias,
                            const float* __restrict__ skip, float* __restrict__ out,
                            int N) {
    int d = blockIdx.x * (blockDim.x >> 5) + (threadIdx.x >> 5);
    int lane = threadIdx.x & 31;
    if (d >= N) return;
    int r0 = rowptr[d], r1 = rowptr[d + 1];

    int hs = lane & 3;                       // stats head for this lane
    float m = -1e30f;
    for (int j = r0 + (lane >> 2); j < r1; j += 8)
        m = fmaxf(m, elog[(size_t)eidx[j] * 4 + hs]);
    m = fmaxf(m, __shfl_xor(m, 4));
    m = fmaxf(m, __shfl_xor(m, 8));
    m = fmaxf(m, __shfl_xor(m, 16));
    float ssum = 0.0f;
    for (int j = r0 + (lane >> 2); j < r1; j += 8)
        ssum += __expf(elog[(size_t)eidx[j] * 4 + hs] - m);
    ssum += __shfl_xor(ssum, 4);
    ssum += __shfl_xor(ssum, 8);
    ssum += __shfl_xor(ssum, 16);

    int h2 = lane >> 3;                      // feature head of this lane
    float mh  = __shfl(m, h2);               // lane h2 holds head h2 stats
    float inv = 1.0f / __shfl(ssum, h2);

    float4 acc = {0.f, 0.f, 0.f, 0.f};
    for (int j = r0; j < r1; j++) {
        int sn = esrc[j];
        float wgt = __expf(elog[(size_t)eidx[j] * 4 + h2] - mh) * inv;
        float4 v = ((const float4*)(xl + (size_t)sn * 128))[lane];
        acc.x += wgt * v.x; acc.y += wgt * v.y;
        acc.z += wgt * v.z; acc.w += wgt * v.w;
    }
    float4 bb = ((const float4*)bias)[lane];
    float4 sk = ((const float4*)(skip + (size_t)d * 128))[lane];
    float4 o;
    o.x = fmaxf(acc.x + bb.x + sk.x, 0.0f);  // fused +b1 +skip1 +relu
    o.y = fmaxf(acc.y + bb.y + sk.y, 0.0f);
    o.z = fmaxf(acc.z + bb.z + sk.z, 0.0f);
    o.w = fmaxf(acc.w + bb.w + sk.w, 0.0f);
    ((float4*)(out + (size_t)d * 128))[lane] = o;
}

// Single-head variant: stats across the whole wave; no relu (layer-2 fusion).
__global__ void k_gather_h1(const float* __restrict__ xl, const float* __restrict__ elog,
                            const int* __restrict__ rowptr, const int* __restrict__ esrc,
                            const int* __restrict__ eidx, const float* __restrict__ bias,
                            const float* __restrict__ skip, float* __restrict__ out,
                            int N) {
    int d = blockIdx.x * (blockDim.x >> 5) + (threadIdx.x >> 5);
    int lane = threadIdx.x & 31;
    if (d >= N) return;
    int r0 = rowptr[d], r1 = rowptr[d + 1];

    float m = -1e30f;
    for (int j = r0 + lane; j < r1; j += 32) m = fmaxf(m, elog[eidx[j]]);
    for (int msk = 1; msk < 32; msk <<= 1) m = fmaxf(m, __shfl_xor(m, msk));
    float ssum = 0.0f;
    for (int j = r0 + lane; j < r1; j += 32) ssum += __expf(elog[eidx[j]] - m);
    for (int msk = 1; msk < 32; msk <<= 1) ssum += __shfl_xor(ssum, msk);
    float inv = 1.0f / ssum;

    float4 acc = {0.f, 0.f, 0.f, 0.f};
    for (int j = r0; j < r1; j++) {
        int sn = esrc[j];
        float wgt = __expf(elog[eidx[j]] - m) * inv;
        float4 v = ((const float4*)(xl + (size_t)sn * 128))[lane];
        acc.x += wgt * v.x; acc.y += wgt * v.y;
        acc.z += wgt * v.z; acc.w += wgt * v.w;
    }
    float4 bb = ((const float4*)bias)[lane];
    float4 sk = ((const float4*)(skip + (size_t)d * 128))[lane];
    float4 o;
    o.x = acc.x + bb.x + sk.x;               // h2 + b2 + (h@skip2_w + skip2_b)
    o.y = acc.y + bb.y + sk.y;
    o.z = acc.z + bb.z + sk.z;
    o.w = acc.w + bb.w + sk.w;
    ((float4*)(out + (size_t)d * 128))[lane] = o;
}

// ------------------------------ host driver --------------------------------

static inline char* wsoff(void* ws, size_t& off, size_t bytes) {
    char* p = (char*)ws + off;
    off += (bytes + 255) & ~(size_t)255;
    return p;
}

extern "C" void kernel_launch(void* const* d_in, const int* in_sizes, int n_in,
                              void* d_out, int out_size, void* d_ws, size_t ws_size,
                              hipStream_t stream) {
    const int N = in_sizes[0] / INF_;          // 100000
    const int E = in_sizes[1] / 2;             // 1600000
    const int ET = E + N;                      // with self loops

    const float* x       = (const float*)d_in[0];
    const int*   ei      = (const int*)d_in[1];   // int32 (JAX x64 disabled)
    const int*   srcp    = ei;
    const int*   dstp    = ei + E;
    const float* Wl1     = (const float*)d_in[2];
    const float* bl1     = (const float*)d_in[3];
    const float* Wr1     = (const float*)d_in[4];
    const float* br1     = (const float*)d_in[5];
    const float* att1    = (const float*)d_in[6];
    const float* b1      = (const float*)d_in[7];
    const float* Wl2     = (const float*)d_in[8];
    const float* bl2     = (const float*)d_in[9];
    const float* Wr2     = (const float*)d_in[10];
    const float* br2     = (const float*)d_in[11];
    const float* att2    = (const float*)d_in[12];
    const float* b2      = (const float*)d_in[13];
    const float* skip1_w = (const float*)d_in[14];
    const float* skip1_b = (const float*)d_in[15];
    const float* skip2_w = (const float*)d_in[16];
    const float* skip2_b = (const float*)d_in[17];
    const float* mlp1_w  = (const float*)d_in[18];
    const float* mlp1_b  = (const float*)d_in[19];
    const float* mlp2_w  = (const float*)d_in[20];
    const float* mlp2_b  = (const float*)d_in[21];
    float* outp = (float*)d_out;

    size_t off = 0;
    _Float16* xh16a = (_Float16*)wsoff(d_ws, off, (size_t)N * 128 * 2);
    _Float16* xh16b = (_Float16*)wsoff(d_ws, off, (size_t)N * 128 * 2);
    _Float16* wt    = (_Float16*)wsoff(d_ws, off, (size_t)128 * 128 * 2);
    float* xl1   = (float*)wsoff(d_ws, off, (size_t)N * 128 * 4);
    float* xr1   = (float*)wsoff(d_ws, off, (size_t)N * 128 * 4);
    float* skipv = (float*)wsoff(d_ws, off, (size_t)N * 128 * 4);
    float* hbuf  = (float*)wsoff(d_ws, off, (size_t)N * 128 * 4);
    float* elog  = (float*)wsoff(d_ws, off, (size_t)ET * 4 * 4);
    int* degfill = (int*)wsoff(d_ws, off, (size_t)2 * N * 4);
    int* deg  = degfill;
    int* fill = degfill + N;
    int* rowptr   = (int*)wsoff(d_ws, off, (size_t)(N + 1) * 4);
    int* partials = (int*)wsoff(d_ws, off, (size_t)1024 * 4);
    int* esrc = (int*)wsoff(d_ws, off, (size_t)ET * 4);
    int* eidx = (int*)wsoff(d_ws, off, (size_t)ET * 4);

    const int B = 256;
    const int wavesPerBlk = B / 32;
    dim3 blk(B);
    int nbScan = (N + SCAN_B - 1) / SCAN_B;

    // ---- CSR build (once, reused by both layers) ----
    k_zero_i32<<<(2 * N + B - 1) / B, blk, 0, stream>>>(degfill, 2 * N);
    k_count_deg<<<(ET + B - 1) / B, blk, 0, stream>>>(dstp, deg, E, N);
    k_scan_block<<<nbScan, dim3(SCAN_B), 0, stream>>>(deg, rowptr, partials, N);
    k_scan_partials<<<1, dim3(1024), 0, stream>>>(partials, nbScan);
    k_scan_add<<<(N + B - 1) / B, blk, 0, stream>>>(rowptr, partials, N);
    k_build_csr<<<(ET + B - 1) / B, blk, 0, stream>>>(srcp, dstp, rowptr, fill,
                                                      esrc, eidx, E, N);

    // ---- layer 1 node transforms (WMMA) ----
    int nv4 = N * 128 / 4;
    k_f2h<<<(nv4 + B - 1) / B, blk, 0, stream>>>(x, xh16a, nv4);
    int tilesM = (N + 15) / 16;
    dim3 gGemm((tilesM + wavesPerBlk - 1) / wavesPerBlk, HIDF / 64);
    dim3 bGemm(B);
    int pw = 128 * 128;
    k_packw<<<(pw + B - 1) / B, blk, 0, stream>>>(Wl1, wt, 128);
    k_gemm_wmma<<<gGemm, bGemm, 0, stream>>>(xh16a, wt, bl1, xl1, nullptr, N, 128, 0);
    k_packw<<<(pw + B - 1) / B, blk, 0, stream>>>(Wr1, wt, 128);
    k_gemm_wmma<<<gGemm, bGemm, 0, stream>>>(xh16a, wt, br1, xr1, nullptr, N, 128, 0);
    k_packw<<<(pw + B - 1) / B, blk, 0, stream>>>(skip1_w, wt, 128);
    k_gemm_wmma<<<gGemm, bGemm, 0, stream>>>(xh16a, wt, skip1_b, skipv, nullptr, N, 128, 0);

    // ---- layer 1 edge stage ----
    int gEdge = (ET + wavesPerBlk - 1) / wavesPerBlk;
    k_elog_h4<<<gEdge, blk, 0, stream>>>(xl1, xr1, srcp, dstp, att1, elog, E, N);
    int gNode = (N + wavesPerBlk - 1) / wavesPerBlk;
    k_gather_h4<<<gNode, blk, 0, stream>>>(xl1, elog, rowptr, esrc, eidx,
                                           b1, skipv, hbuf, N);

    // ---- layer 2 node transforms ----
    k_f2h<<<(nv4 + B - 1) / B, blk, 0, stream>>>(hbuf, xh16a, nv4);
    k_packw<<<(pw + B - 1) / B, blk, 0, stream>>>(Wl2, wt, 128);
    k_gemm_wmma<<<gGemm, bGemm, 0, stream>>>(xh16a, wt, bl2, xl1, nullptr, N, 128, 0);
    k_packw<<<(pw + B - 1) / B, blk, 0, stream>>>(Wr2, wt, 128);
    k_gemm_wmma<<<gGemm, bGemm, 0, stream>>>(xh16a, wt, br2, xr1, nullptr, N, 128, 0);
    k_packw<<<(pw + B - 1) / B, blk, 0, stream>>>(skip2_w, wt, 128);
    k_gemm_wmma<<<gGemm, bGemm, 0, stream>>>(xh16a, wt, skip2_b, skipv, nullptr, N, 128, 0);

    // ---- layer 2 edge stage (writes over hbuf: h2 + skip2 fused) ----
    k_elog_h1<<<gEdge, blk, 0, stream>>>(xl1, xr1, srcp, dstp, att2, elog, E, N);
    k_gather_h1<<<gNode, blk, 0, stream>>>(xl1, elog, rowptr, esrc, eidx,
                                           b2, skipv, hbuf, N);

    // ---- MLP head ----
    k_f2h<<<(nv4 + B - 1) / B, blk, 0, stream>>>(hbuf, xh16a, nv4);
    k_packw<<<(pw + B - 1) / B, blk, 0, stream>>>(mlp1_w, wt, 128);
    k_gemm_wmma<<<gGemm, bGemm, 0, stream>>>(xh16a, wt, mlp1_b, nullptr, xh16b,
                                             N, 128, 1);          // relu, f16 out
    int pw2 = 128 * 64;
    k_packw<<<(pw2 + B - 1) / B, blk, 0, stream>>>(mlp2_w, wt, 64);
    dim3 gGemm2((tilesM + wavesPerBlk - 1) / wavesPerBlk, OUTF / 64);
    k_gemm_wmma<<<gGemm2, bGemm, 0, stream>>>(xh16b, wt, mlp2_b, outp, nullptr,
                                              N, 64, 0);
    (void)ws_size; (void)n_in; (void)out_size;
}
